// DSTP_RNN_II_11690900980423
// MI455X (gfx1250) — matrix operand order repaired and checked by reading, not compiled, not run on hardware
//
#include <hip/hip_runtime.h>

typedef _Float16 h16;
typedef __attribute__((ext_vector_type(16))) _Float16 v16h;
typedef __attribute__((ext_vector_type(8)))  float    v8f;

#define BLK 128   // 4 waves of 32

__device__ __forceinline__ v8f wmma16(v16h a, v16h b, v8f c) {
  // D = A(16x32 f16) * B(32x16 f16) + C(16x16 f32)
  return __builtin_amdgcn_wmma_f32_16x16x32_f16(false, a, false, b, (short)0, c,
                                                false, false);
}
__device__ __forceinline__ float sigm(float x) { return 1.f / (1.f + __expf(-x)); }

// A-matrix (16x32 f16) VGPR layout per CDNA5 ISA 7.12.2:
// lane = m + 16*((kk>>3)&1);  elem = (kk&7) + (kk>=16 ? 8 : 0)
// LDS slab per k-chunk: 32 lanes * 16 halfs contiguous (512 halfs / 1KB).
__device__ __forceinline__ int a_idx(int k, int m) {
  int kc = k >> 5, kk = k & 31;
  int lane = m + (((kk >> 3) & 1) << 4);
  int e = (kk & 7) + ((kk & 16) ? 8 : 0);
  return kc * 512 + lane * 16 + e;
}

// ---------------------------------------------------------------------------
// Prep: pack weights into f16 WMMA-B fragment layout.
// B(32x16) layout: lane = n_local + 16*((k_local>>3)&1); elem = (k_local&7) + (k_local>=16 ? 8 : 0).
// Buffer layout: frag(nt,kc) at ((nt*KC+kc)*32 + lane)*16 halfs.
// Logical B[k][n] = k < in_d         ? Wih[n*in_d + k]
//                 : (Whh && k<in_d+128)? Whh[n*128 + (k-in_d)] : 0
// ---------------------------------------------------------------------------
__global__ void swizzleB(const float* __restrict__ Wih, const float* __restrict__ Whh,
                         h16* __restrict__ dst, int in_d, int N, int KC) {
  int total = (N / 16) * KC * 512;
  for (int idx = blockIdx.x * blockDim.x + threadIdx.x; idx < total;
       idx += gridDim.x * blockDim.x) {
    int e = idx & 15;
    int lane = (idx >> 4) & 31;
    int kc = (idx >> 9) % KC;
    int nt = idx / (KC * 512);
    int n = nt * 16 + (lane & 15);
    int k = kc * 32 + ((lane >> 4) << 3) + (e & 7) + ((e >> 3) << 4);
    float v = 0.f;
    if (k < in_d) v = Wih[(size_t)n * in_d + k];
    else if (Whh != nullptr && k < in_d + 128) v = Whh[(size_t)n * 128 + (k - in_d)];
    dst[idx] = (h16)v;
  }
}

// ---------------------------------------------------------------------------
// Feature fetch for the three spatial encoders.
// mode 0: input_p_q cols {0..13,15..17} (in_d=17)
// mode 1: input_p_q all 18 cols
// mode 2: concat(mid11 f16, mid12 f16, label f32) (in_d=257)
// ---------------------------------------------------------------------------
__device__ __forceinline__ float fetch_x(int mode, const float* inp, const h16* m11,
                                         const h16* m12, const float* label,
                                         long b, int t, int n) {
  if (mode == 0) { int col = (n < 14) ? n : n + 1; return inp[(b * 64 + t) * 18 + col]; }
  if (mode == 1) return inp[(b * 64 + t) * 18 + n];
  if (n < 128)  return (float)m11[(b * 64 + t) * 128 + n];
  if (n < 256)  return (float)m12[(b * 64 + t) * 128 + (n - 128)];
  return label[b * 64 + t];
}

// ---------------------------------------------------------------------------
// Spatial encoder: one WG = 16 batch rows, full 64-step scan.
// LDS (dynamic, 62464 B):
//   A_hc   @ 0      (8 KB)   f16 A-layout of [h(128), c(128)]      (K=256)
//   A_xh   @ 8192   (13 KB)  f16 A-layout of [x*w(in_d), h(128), pad] (K<=416)
//   R      @ 21504  (32 KB)  union{ gates f32 16x512 | scores 16x260 + we 16x64 }
//   cS     @ 54272  (8 KB)   f32 c state 16x128
// ---------------------------------------------------------------------------
__global__ void __launch_bounds__(BLK) encoder_kernel(
    const float* __restrict__ inp, const h16* __restrict__ m11,
    const h16* __restrict__ m12, const float* __restrict__ label,
    const float* __restrict__ Wi, const float* __restrict__ Wi_b,
    const float* __restrict__ Vd, const float* __restrict__ Vd_b,
    const h16* __restrict__ WeT, const h16* __restrict__ Wcat,
    const float* __restrict__ b_ih, const float* __restrict__ b_hh,
    h16* __restrict__ wi_buf, h16* __restrict__ hs_out, int in_d, int KC, int mode) {
  extern __shared__ char smem[];
  const int O_AHC = 0, O_AXH = 8192, O_R = 21504, O_C = 54272;
  h16* A_hc = (h16*)(smem + O_AHC);
  h16* A_xh = (h16*)(smem + O_AXH);
  float* wrow = (float*)(smem + O_R);              // 16 x 260 scores/softmax
  float* weS = (float*)(smem + O_R + 16640);       // 16 x 64
  float* gatesS = (float*)(smem + O_R);            // 16 x 512 (aliases wrow/weS)
  float* cS = (float*)(smem + O_C);                // 16 x 128

  const int tid = threadIdx.x, lane = tid & 31, wave = tid >> 5;
  const long b0 = (long)blockIdx.x * 16;
  const v8f zero8 = {0.f, 0.f, 0.f, 0.f, 0.f, 0.f, 0.f, 0.f};

  // zero state + A staging (pad region of A_xh stays zero forever)
  {
    int* z0 = (int*)(smem + O_AHC);
    for (int i = tid; i < 2048; i += BLK) z0[i] = 0;
    int* z1 = (int*)(smem + O_AXH);
    for (int i = tid; i < KC * 256; i += BLK) z1[i] = 0;
    for (int i = tid; i < 2048; i += BLK) cS[i] = 0.f;
  }
  __syncthreads();

  // ---- wi[m][n][s] = sum_t x(m,t,n) * Wi[s][t] + Wi_b[s]  (step-invariant) ----
  for (int p = tid; p < 16 * in_d; p += BLK) {
    int m = p / in_d, n = p - m * in_d;
    float xv[64];
#pragma unroll
    for (int t = 0; t < 64; ++t) xv[t] = fetch_x(mode, inp, m11, m12, label, b0 + m, t, n);
    h16* dstp = wi_buf + (((size_t)(b0 + m)) * in_d + n) * 64;
#pragma unroll 1
    for (int s = 0; s < 64; ++s) {
      float acc = Wi_b[s];
      const float* wr = Wi + s * 64;
#pragma unroll
      for (int t = 0; t < 64; ++t) acc += xv[t] * wr[t];
      dstp[s] = (h16)acc;
    }
  }
  __threadfence();
  __syncthreads();

  // ------------------------------ 64-step scan ------------------------------
  for (int t = 0; t < 64; ++t) {
    // we = [h,c] @ We^T : 16x64, K=256 (KC=8). One N-tile per wave.
    {
      v8f acc = zero8;
#pragma unroll
      for (int kc = 0; kc < 8; ++kc) {
        v16h a = *(const v16h*)(A_hc + kc * 512 + lane * 16);
        v16h bf = *(const v16h*)(WeT + ((size_t)(wave * 8 + kc) * 32 + lane) * 16);
        acc = wmma16(a, bf, acc);
      }
      int mhi = (lane >> 4) * 8, nl = lane & 15;
#pragma unroll
      for (int r = 0; r < 8; ++r) weS[(mhi + r) * 64 + wave * 16 + nl] = acc[r];
    }
    __syncthreads();

    // score[m][n] = sum_s tanh(wi + we) * Vd[s] + Vd_b
    for (int p = tid; p < 16 * in_d; p += BLK) {
      int m = p / in_d, n = p - m * in_d;
      const h16* wip = wi_buf + (((size_t)(b0 + m)) * in_d + n) * 64;
      const float* wem = weS + m * 64;
      float acc = 0.f;
#pragma unroll 4
      for (int s = 0; s < 64; ++s) acc += tanhf((float)wip[s] + wem[s]) * Vd[s];
      wrow[m * 260 + n] = acc + Vd_b[0];
    }
    __syncthreads();

    // softmax over features (one row per thread)
    if (tid < 16) {
      int m = tid;
      float mx = -1e30f;
      for (int n = 0; n < in_d; ++n) mx = fmaxf(mx, wrow[m * 260 + n]);
      float s = 0.f;
      for (int n = 0; n < in_d; ++n) {
        float ev = __expf(wrow[m * 260 + n] - mx);
        wrow[m * 260 + n] = ev; s += ev;
      }
      float inv = 1.f / s;
      for (int n = 0; n < in_d; ++n) wrow[m * 260 + n] *= inv;
    }
    __syncthreads();

    // gated input -> A_xh (A-layout f16), k = 0..in_d-1
    for (int p = tid; p < 16 * in_d; p += BLK) {
      int m = p / in_d, n = p - m * in_d;
      float xv = fetch_x(mode, inp, m11, m12, label, b0 + m, t, n) * wrow[m * 260 + n];
      A_xh[a_idx(n, m)] = (h16)xv;
    }
    __syncthreads();

    // gates = [x*w, h] @ [W_ih; W_hh]^T : 16x512.  8 N-tiles per wave.
    v8f g[8];
#pragma unroll
    for (int j = 0; j < 8; ++j) g[j] = zero8;
    for (int kc = 0; kc < KC; ++kc) {
      v16h a = *(const v16h*)(A_xh + kc * 512 + lane * 16);
      if (kc + 1 < KC)
        __builtin_prefetch(Wcat + ((size_t)(wave * 8) * KC + kc + 1) * 512, 0, 1);
#pragma unroll
      for (int j = 0; j < 8; ++j) {
        size_t nt = (size_t)(wave * 8 + j);
        v16h bf = *(const v16h*)(Wcat + ((nt * KC + kc) * 32 + lane) * 16);
        g[j] = wmma16(a, bf, g[j]);
      }
    }
    {
      int mhi = (lane >> 4) * 8, nl = lane & 15;
      for (int j = 0; j < 8; ++j) {
        int nb = (wave * 8 + j) * 16 + nl;
#pragma unroll
        for (int r = 0; r < 8; ++r) gatesS[(mhi + r) * 512 + nb] = g[j][r];
      }
    }
    __syncthreads();

    // LSTM nonlinearity; restage h,c into f16 A-layout; emit h
    for (int p = tid; p < 2048; p += BLK) {
      int m = p >> 7, k = p & 127;
      const float* gm = gatesS + m * 512;
      float gi = gm[k]       + b_ih[k]       + b_hh[k];
      float gf = gm[128 + k] + b_ih[128 + k] + b_hh[128 + k];
      float gg = gm[256 + k] + b_ih[256 + k] + b_hh[256 + k];
      float go = gm[384 + k] + b_ih[384 + k] + b_hh[384 + k];
      float cn = sigm(gf) * cS[p] + sigm(gi) * tanhf(gg);
      float hn = sigm(go) * tanhf(cn);
      cS[p] = cn;
      A_hc[a_idx(k, m)] = (h16)hn;
      A_hc[a_idx(128 + k, m)] = (h16)cn;
      A_xh[a_idx(in_d + k, m)] = (h16)hn;
      hs_out[(((size_t)(b0 + m)) * 64 + t) * 128 + k] = (h16)hn;
    }
    __syncthreads();
  }
}

// ---------------------------------------------------------------------------
// wx = final @ Wx^T + b : (B*64, 128) rows, K=128 (KC=4). WG = 16 rows.
// ---------------------------------------------------------------------------
__global__ void __launch_bounds__(BLK) wx_kernel(const h16* __restrict__ finalb,
                                                 const h16* __restrict__ WxT,
                                                 const float* __restrict__ Wx_b,
                                                 h16* __restrict__ wx) {
  extern __shared__ char smem[];
  h16* A = (h16*)smem;  // 4 k-chunk slabs = 4 KB
  int tid = threadIdx.x, lane = tid & 31, wave = tid >> 5;
  size_t r0 = (size_t)blockIdx.x * 16;
  for (int p = tid; p < 16 * 128; p += BLK) {
    int m = p >> 7, k = p & 127;
    A[a_idx(k, m)] = finalb[(r0 + m) * 128 + k];
  }
  __syncthreads();
  const v8f zero8 = {0.f, 0.f, 0.f, 0.f, 0.f, 0.f, 0.f, 0.f};
  for (int j = 0; j < 2; ++j) {
    int nt = wave * 2 + j;
    v8f acc = zero8;
#pragma unroll
    for (int kc = 0; kc < 4; ++kc) {
      v16h a = *(const v16h*)(A + kc * 512 + lane * 16);
      v16h bf = *(const v16h*)(WxT + ((size_t)(nt * 4 + kc) * 32 + lane) * 16);
      acc = wmma16(a, bf, acc);
    }
    int mhi = (lane >> 4) * 8, nl = lane & 15, n = nt * 16 + nl;
    float bias = Wx_b[n];
#pragma unroll
    for (int r = 0; r < 8; ++r) wx[(r0 + mhi + r) * 128 + n] = (h16)(acc[r] + bias);
  }
}

// ---------------------------------------------------------------------------
// Decoder: 30 steps of temporal attention + LSTM. WG = 16 batch rows.
// LDS (57344 B): A_hc@0 (8K, [h,c]) | A_dh@8192 (8K, [din,h]) |
//   R@16384 union{gates 32K | wrow 16x64 + wh 16x128} | cS@49152 (8K)
// ---------------------------------------------------------------------------
__global__ void __launch_bounds__(BLK) decoder_kernel(
    const h16* __restrict__ finalb, const h16* __restrict__ wxb,
    const h16* __restrict__ WhT, const h16* __restrict__ Wcat,
    const float* __restrict__ V, const float* __restrict__ V_b,
    const float* __restrict__ b_ih, const float* __restrict__ b_hh,
    const float* __restrict__ Wreg, const float* __restrict__ breg,
    float* __restrict__ out) {
  extern __shared__ char smem[];
  const int O_AHC = 0, O_ADH = 8192, O_R = 16384, O_C = 49152;
  h16* A_hc = (h16*)(smem + O_AHC);
  h16* A_dh = (h16*)(smem + O_ADH);
  float* wrow = (float*)(smem + O_R);         // 16 x 64
  float* whS = (float*)(smem + O_R + 4096);   // 16 x 128
  float* gatesS = (float*)(smem + O_R);       // 16 x 512 (aliases the above)
  float* cS = (float*)(smem + O_C);
  int tid = threadIdx.x, lane = tid & 31, wave = tid >> 5;
  long b0 = (long)blockIdx.x * 16;
  const v8f zero8 = {0.f, 0.f, 0.f, 0.f, 0.f, 0.f, 0.f, 0.f};

  {
    int* z = (int*)smem;
    for (int i = tid; i < 4096; i += BLK) z[i] = 0;  // A_hc + A_dh
    for (int i = tid; i < 2048; i += BLK) cS[i] = 0.f;
  }
  __syncthreads();

  for (int tt = 0; tt < 30; ++tt) {
    // wh = [h,c] @ Wh^T : 16x128, K=256 (KC=8). 2 N-tiles per wave.
    for (int j = 0; j < 2; ++j) {
      v8f acc = zero8;
#pragma unroll
      for (int kc = 0; kc < 8; ++kc) {
        v16h a = *(const v16h*)(A_hc + kc * 512 + lane * 16);
        v16h bf = *(const v16h*)(WhT + ((size_t)((wave * 2 + j) * 8 + kc) * 32 + lane) * 16);
        acc = wmma16(a, bf, acc);
      }
      int mhi = (lane >> 4) * 8, nl = lane & 15, n = (wave * 2 + j) * 16 + nl;
#pragma unroll
      for (int r = 0; r < 8; ++r) whS[(mhi + r) * 128 + n] = acc[r];
    }
    __syncthreads();

    // score[m][t2] = sum_h tanh(wx + wh) * V[h] + V_b
    for (int p = tid; p < 16 * 64; p += BLK) {
      int m = p >> 6, t2 = p & 63;
      const h16* wxp = wxb + ((b0 + m) * 64 + t2) * 128;
      const float* whm = whS + m * 128;
      float acc = 0.f;
#pragma unroll 4
      for (int h = 0; h < 128; ++h) acc += tanhf((float)wxp[h] + whm[h]) * V[h];
      wrow[p] = acc + V_b[0];
    }
    __syncthreads();

    if (tid < 16) {
      int m = tid;
      float mx = -1e30f;
      for (int q = 0; q < 64; ++q) mx = fmaxf(mx, wrow[m * 64 + q]);
      float s = 0.f;
      for (int q = 0; q < 64; ++q) {
        float ev = __expf(wrow[m * 64 + q] - mx);
        wrow[m * 64 + q] = ev; s += ev;
      }
      float inv = 1.f / s;
      for (int q = 0; q < 64; ++q) wrow[m * 64 + q] *= inv;
    }
    __syncthreads();

    // din[m][k] = sum_t w[m][t] * final[b][t][k] -> A_dh (k = 0..127)
    for (int p = tid; p < 2048; p += BLK) {
      int m = p >> 7, k = p & 127;
      const float* wm = wrow + m * 64;
      float acc = 0.f;
#pragma unroll 4
      for (int q = 0; q < 64; ++q)
        acc += wm[q] * (float)finalb[((b0 + m) * 64 + q) * 128 + k];
      A_dh[a_idx(k, m)] = (h16)acc;
    }
    __syncthreads();

    // gates = [din, h] @ [W_ih; W_hh]^T : 16x512, K=256 (KC=8)
    v8f g[8];
#pragma unroll
    for (int j = 0; j < 8; ++j) g[j] = zero8;
    for (int kc = 0; kc < 8; ++kc) {
      v16h a = *(const v16h*)(A_dh + kc * 512 + lane * 16);
#pragma unroll
      for (int j = 0; j < 8; ++j) {
        v16h bf = *(const v16h*)(Wcat + ((size_t)((wave * 8 + j) * 8 + kc) * 32 + lane) * 16);
        g[j] = wmma16(a, bf, g[j]);
      }
    }
    {
      int mhi = (lane >> 4) * 8, nl = lane & 15;
      for (int j = 0; j < 8; ++j) {
        int nb = (wave * 8 + j) * 16 + nl;
#pragma unroll
        for (int r = 0; r < 8; ++r) gatesS[(mhi + r) * 512 + nb] = g[j][r];
      }
    }
    __syncthreads();

    for (int p = tid; p < 2048; p += BLK) {
      int m = p >> 7, k = p & 127;
      const float* gm = gatesS + m * 512;
      float gi = gm[k]       + b_ih[k]       + b_hh[k];
      float gf = gm[128 + k] + b_ih[128 + k] + b_hh[128 + k];
      float gg = gm[256 + k] + b_ih[256 + k] + b_hh[256 + k];
      float go = gm[384 + k] + b_ih[384 + k] + b_hh[384 + k];
      float cn = sigm(gf) * cS[p] + sigm(gi) * tanhf(gg);
      float hn = sigm(go) * tanhf(cn);
      cS[p] = cn;
      A_hc[a_idx(k, m)] = (h16)hn;
      A_hc[a_idx(128 + k, m)] = (h16)cn;
      A_dh[a_idx(128 + k, m)] = (h16)hn;
    }
    __syncthreads();

    // out[b][tt-6] = h . Wreg + breg   (last 24 of 30 steps)
    if (tt >= 6 && tid < 16) {
      int m = tid;
      float acc = breg[0];
      for (int k = 0; k < 128; ++k) acc += (float)A_hc[a_idx(k, m)] * Wreg[k];
      out[(b0 + m) * 24 + (tt - 6)] = acc;
    }
    __syncthreads();
  }
}

// ---------------------------------------------------------------------------
extern "C" void kernel_launch(void* const* d_in, const int* in_sizes, int n_in,
                              void* d_out, int out_size, void* d_ws, size_t ws_size,
                              hipStream_t stream) {
  (void)in_sizes; (void)n_in; (void)out_size; (void)ws_size;
  // d_in order (setup_inputs dict order, params flattened depth-first):
  const float* input_p_q = (const float*)d_in[0];
  const float* label_p = (const float*)d_in[1];
  // att11: Wi, Wi_b, We, Vd, Vd_b  -> 2..6 ; att12 -> 7..11 ; att2 -> 12..16
  // cell11: W_ih,W_hh,b_ih,b_hh -> 17..20 ; cell12 -> 21..24 ; cell2 -> 25..28
  // dec_cell -> 29..32 ; temp: Wx,Wx_b,Wh,V,V_b -> 33..37 ; reg: W,b -> 38..39
  #define IN(i) ((const float*)d_in[(i)])

  // Workspace layout (all 256B-aligned by construction)
  char* W = (char*)d_ws;
  size_t off = 0;
  auto take = [&](size_t bytes) { char* p = W + off; off += bytes; return p; };
  h16* wcat11 = (h16*)take((size_t)32 * 5 * 512 * 2);    // K_pad=160
  h16* wcat12 = (h16*)take((size_t)32 * 5 * 512 * 2);    // K_pad=160
  h16* wcat2  = (h16*)take((size_t)32 * 13 * 512 * 2);   // K_pad=416
  h16* wcatd  = (h16*)take((size_t)32 * 8 * 512 * 2);    // K_pad=256
  h16* WeT11  = (h16*)take((size_t)4 * 8 * 512 * 2);
  h16* WeT12  = (h16*)take((size_t)4 * 8 * 512 * 2);
  h16* WeT2   = (h16*)take((size_t)4 * 8 * 512 * 2);
  h16* WhT    = (h16*)take((size_t)8 * 8 * 512 * 2);
  h16* WxT    = (h16*)take((size_t)8 * 4 * 512 * 2);
  h16* wi11   = (h16*)take((size_t)2048 * 17 * 64 * 2);
  h16* wi12   = (h16*)take((size_t)2048 * 18 * 64 * 2);
  h16* wi2    = (h16*)take((size_t)2048 * 257 * 64 * 2);
  h16* mid11  = (h16*)take((size_t)2048 * 64 * 128 * 2);
  h16* mid12  = (h16*)take((size_t)2048 * 64 * 128 * 2);
  h16* finalb = (h16*)take((size_t)2048 * 64 * 128 * 2);
  h16* wxb    = (h16*)take((size_t)2048 * 64 * 128 * 2);

  // ---- weight prep ----
  auto sw = [&](const float* wih, const float* whh, h16* dst, int ind, int N, int KC) {
    int total = (N / 16) * KC * 512;
    swizzleB<<<(total + 255) / 256, 256, 0, stream>>>(wih, whh, dst, ind, N, KC);
  };
  sw(IN(17), IN(18), wcat11, 17, 512, 5);
  sw(IN(21), IN(22), wcat12, 18, 512, 5);
  sw(IN(25), IN(26), wcat2, 257, 512, 13);
  sw(IN(29), IN(30), wcatd, 128, 512, 8);
  sw(IN(4), nullptr, WeT11, 256, 64, 8);
  sw(IN(9), nullptr, WeT12, 256, 64, 8);
  sw(IN(14), nullptr, WeT2, 256, 64, 8);
  sw(IN(35), nullptr, WhT, 256, 128, 8);
  sw(IN(33), nullptr, WxT, 128, 128, 4);

  const int GRID = 2048 / 16;
  const size_t ENC_SMEM = 62464, DEC_SMEM = 57344;

  // ---- encoders ----
  encoder_kernel<<<GRID, BLK, ENC_SMEM, stream>>>(
      input_p_q, nullptr, nullptr, nullptr, IN(2), IN(3), IN(5), IN(6),
      WeT11, wcat11, IN(19), IN(20), wi11, mid11, 17, 5, 0);
  encoder_kernel<<<GRID, BLK, ENC_SMEM, stream>>>(
      input_p_q, nullptr, nullptr, nullptr, IN(7), IN(8), IN(10), IN(11),
      WeT12, wcat12, IN(23), IN(24), wi12, mid12, 18, 5, 1);
  encoder_kernel<<<GRID, BLK, ENC_SMEM, stream>>>(
      input_p_q, mid11, mid12, label_p, IN(12), IN(13), IN(15), IN(16),
      WeT2, wcat2, IN(27), IN(28), wi2, finalb, 257, 13, 2);

  // ---- wx = final @ Wx^T + b ----
  wx_kernel<<<(2048 * 64) / 16, BLK, 4096, stream>>>(finalb, WxT, IN(34), wxb);

  // ---- decoder ----
  decoder_kernel<<<GRID, BLK, DEC_SMEM, stream>>>(
      finalb, wxb, WhT, wcatd, IN(36), IN(37), IN(31), IN(32), IN(38), IN(39),
      (float*)d_out);
  #undef IN
}
